// LIFNet_62397284876288
// MI455X (gfx1250) — matrix operand order repaired and checked by reading, not compiled, not run on hardware
//
#include <hip/hip_runtime.h>

// ---------------------------------------------------------------------------
// Fused LIF network for gfx1250 (MI455X), wave32 + WMMA.
//   z1 = batch @ W1^T            [B,T,100]   (bf16 WMMA, f32 accum)
//   V1 = leaky(z1)  (a=0.7,b=0.3)
//   z2 = V1 @ W2^T               [B,T,10]    (tiny f32 matvec)
//   V2 = leaky(z2)  -> out
//
// Single pass over batch (627 MB) — no z1 materialization.
// W1 lives as bf16 in LDS (177 KB of the 320 KB CDNA5 LDS): hot loop is
// branch-free ds_load_b128 -> v_wmma. M tile = 32 timesteps so each wave
// runs two accumulator chains sharing one B fragment (3 ds_loads / wmma).
// T split into 4 segments with 96-step warm-up (alpha^96 ~ 1e-15) to raise
// block parallelism from 100 to 400.
// ---------------------------------------------------------------------------

typedef __attribute__((ext_vector_type(16))) __bf16 v16bf;
typedef __attribute__((ext_vector_type(8)))  __bf16 v8bf;
typedef __attribute__((ext_vector_type(4)))  __bf16 v4bf;
typedef __attribute__((ext_vector_type(8)))  float  v8f;

#define B_      100
#define T_      2000
#define DIN_    784
#define H1_     100
#define H2_     10
#define ALPHA_  0.7f
#define BETA_   0.3f

#define MT      32          // timesteps per chunk (two 16-row M tiles)
#define KSTEPS  25          // ceil(784/32); K zero-padded to 800
#define ASTRIDE 808         // bf16 elements per A row in LDS (conflict pad)
#define WSTRIDE 808         // bf16 elements per W1 row in LDS (conflict pad)
#define NPAD    112         // H1 padded to 7*16
#define ZSTRIDE 120         // f32 elements per Z row in LDS (conflict pad)
#define NT      7           // N tiles of 16 covering H1=100
#define SEGLEN  512         // T segment length (32-aligned)
#define WARM    96          // warm-up steps recomputed per segment (3 chunks)

static __device__ __forceinline__ v16bf ldfrag(const __bf16* p0, const __bf16* p1) {
    const v8bf a = *(const v8bf*)p0;
    const v8bf b = *(const v8bf*)p1;
    return __builtin_shufflevector(a, b, 0,1,2,3,4,5,6,7,8,9,10,11,12,13,14,15);
}

__global__ __launch_bounds__(256) void lif_fused_kernel(
    const float* __restrict__ batch,   // [B,T,DIN]
    const float* __restrict__ W1,      // [H1,DIN]
    const float* __restrict__ W2,      // [H2,H1]
    float* __restrict__ out)           // [B,T,H2]
{
    __shared__ __align__(16) __bf16 W1s[NPAD * WSTRIDE];  // 180992 B (bf16 W1, K-padded)
    __shared__ __align__(16) __bf16 Abuf[MT * ASTRIDE];   //  51712 B
    __shared__ __align__(16) float  Zbuf[MT * ZSTRIDE];   //  15360 B
    __shared__ __align__(16) float  W2s[H2_ * H1_];       //   4000 B
    __shared__ __align__(16) float  V2b[MT * H2_];        //   1280 B

    const int tid  = threadIdx.x;
    const int w    = tid >> 5;        // wave id (wave32)
    const int lane = tid & 31;
    const int half = lane >> 4;
    const int mrow = lane & 15;

    const int seg = blockIdx.x;       // 0..3
    const int b   = blockIdx.y;       // 0..99
    const int seg_start = seg * SEGLEN;
    const int seg_end   = (seg_start + SEGLEN < T_) ? (seg_start + SEGLEN) : T_;
    const int t_begin   = (seg == 0) ? 0 : (seg_start - WARM);

    const float* bb = batch + (size_t)b * T_ * DIN_;

    // ---- One-time: W2 into LDS; W1 -> bf16 into LDS (K zero-padded to 808,
    //      rows >= 100 clamped: they feed only unused output columns).
    for (int i = tid; i < H2_ * H1_; i += 256) W2s[i] = W2[i];

    for (int g = tid; g < NPAD * (WSTRIDE / 4); g += 256) {
        const int n  = g / (WSTRIDE / 4);
        const int kg = (g - n * (WSTRIDE / 4)) * 4;
        const int ns = (n < H1_) ? n : (H1_ - 1);
        v4bf v;
        if (kg < DIN_) {
            const float4 f = *(const float4*)(W1 + (size_t)ns * DIN_ + kg);
            v[0] = (__bf16)f.x; v[1] = (__bf16)f.y;
            v[2] = (__bf16)f.z; v[3] = (__bf16)f.w;
        } else {
            v[0] = (__bf16)0.0f; v[1] = (__bf16)0.0f;
            v[2] = (__bf16)0.0f; v[3] = (__bf16)0.0f;
        }
        *(v4bf*)&W1s[n * WSTRIDE + kg] = v;
    }

    // Per-thread leaky-integrator state (thread<->channel mapping fixed).
    float S1 = 0.0f;   // used by tid < 100 (h = tid)
    float S2 = 0.0f;   // used by tid < 10  (o = tid)

    __syncthreads();

    for (int t0 = t_begin; t0 < seg_end; t0 += MT) {
        // ---- Stage A tile: 32 rows (t0..t0+31) x K, fp32 -> bf16.
        //      Zero-pad K >= 784 and rows past T (tail chunk).
        for (int g = tid; g < MT * (ASTRIDE / 4); g += 256) {
            const int i  = g / (ASTRIDE / 4);
            const int kg = (g - i * (ASTRIDE / 4)) * 4;
            v4bf v;
            if (kg < DIN_ && (t0 + i) < T_) {
                const float4 f = *(const float4*)(bb + (size_t)(t0 + i) * DIN_ + kg);
                v[0] = (__bf16)f.x; v[1] = (__bf16)f.y;
                v[2] = (__bf16)f.z; v[3] = (__bf16)f.w;
            } else {
                v[0] = (__bf16)0.0f; v[1] = (__bf16)0.0f;
                v[2] = (__bf16)0.0f; v[3] = (__bf16)0.0f;
            }
            *(v4bf*)&Abuf[i * ASTRIDE + kg] = v;
        }
        // Prefetch next chunk's batch rows toward L2/WGP$ (clamped in-range).
        if (t0 + MT < seg_end) {
            int prow = t0 + MT + (tid & 31);
            prow = (prow < T_) ? prow : (T_ - 1);
            __builtin_prefetch(bb + (size_t)prow * DIN_ + (tid >> 5) * 98, 0, 3);
        }
        __syncthreads();   // (1) A ready

        // ---- GEMM1: waves 0..6 each compute a 32x16 D tile over K=800.
        //      B fragment shared by two M tiles: 6 ds_loads per 2 WMMAs.
        if (w < NT) {
            const int ntile = w;
            const int ncol  = ntile * 16 + mrow;             // output channel h
            const __bf16* arow0 = &Abuf[mrow * ASTRIDE];          // rows 0..15
            const __bf16* arow1 = &Abuf[(16 + mrow) * ASTRIDE];   // rows 16..31
            const __bf16* brow  = &W1s[ncol * WSTRIDE];

            v8f acc0 = {0.f, 0.f, 0.f, 0.f, 0.f, 0.f, 0.f, 0.f};
            v8f acc1 = {0.f, 0.f, 0.f, 0.f, 0.f, 0.f, 0.f, 0.f};

            #pragma unroll
            for (int kt = 0; kt < KSTEPS; ++kt) {
                const int base = kt * 32;
                const v16bf Bf  = ldfrag(brow  + base + half * 8,
                                         brow  + base + 16 + half * 8);
                const v16bf Af0 = ldfrag(arow0 + base + half * 8,
                                         arow0 + base + 16 + half * 8);
                const v16bf Af1 = ldfrag(arow1 + base + half * 8,
                                         arow1 + base + 16 + half * 8);
                acc0 = __builtin_amdgcn_wmma_f32_16x16x32_bf16(
                    false, Af0, false, Bf, (short)0, acc0, false, false);
                acc1 = __builtin_amdgcn_wmma_f32_16x16x32_bf16(
                    false, Af1, false, Bf, (short)0, acc1, false, false);
            }

            // D layout: VGPR r -> (M=r, N=lane) lanes 0-15; (M=8+r, N=lane-16).
            #pragma unroll
            for (int r = 0; r < 8; ++r) {
                Zbuf[(half * 8 + r) * ZSTRIDE + ntile * 16 + mrow]        = acc0[r];
                Zbuf[(16 + half * 8 + r) * ZSTRIDE + ntile * 16 + mrow]   = acc1[r];
            }
        }
        __syncthreads();   // (2) z1 tile ready

        // ---- Leaky integrate layer 1, in place: Zbuf becomes V1 tile.
        if (tid < H1_) {
            float* col = &Zbuf[tid];
            #pragma unroll
            for (int i = 0; i < MT; ++i) {
                const float z = col[i * ZSTRIDE];
                col[i * ZSTRIDE] = S1;
                S1 = ALPHA_ * S1 + BETA_ * z;
            }
        }
        __syncthreads();   // (3) V1 ready

        // ---- z2[t][o] = sum_h W2[o][h] * V1[t][h]   (320 dots over 256 thr)
        for (int idx = tid; idx < MT * H2_; idx += 256) {
            const int t = idx / H2_;
            const int o = idx - t * H2_;
            const float* w2row = &W2s[o * H1_];
            const float* vrow  = &Zbuf[t * ZSTRIDE];
            float a2 = 0.0f;
            #pragma unroll 4
            for (int h = 0; h < H1_; ++h) a2 = fmaf(w2row[h], vrow[h], a2);
            V2b[idx] = a2;
        }
        __syncthreads();   // (4) z2 ready

        // ---- Leaky integrate layer 2, in place: V2b becomes V2 tile.
        if (tid < H2_) {
            float* col = &V2b[tid];
            #pragma unroll
            for (int i = 0; i < MT; ++i) {
                const float z = col[i * H2_];
                col[i * H2_] = S2;
                S2 = ALPHA_ * S2 + BETA_ * z;
            }
        }
        __syncthreads();   // (5) V2 ready

        // ---- Coalesced store (skip warm-up region; clamp tail chunk).
        if (t0 >= seg_start) {
            const int nvalid = ((seg_end - t0) < MT ? (seg_end - t0) : MT) * H2_;
            for (int idx = tid; idx < nvalid; idx += 256)
                out[(size_t)b * (T_ * H2_) + (size_t)t0 * H2_ + idx] = V2b[idx];
        }
        __syncthreads();   // (6) protect LDS for next chunk
    }
}

extern "C" void kernel_launch(void* const* d_in, const int* in_sizes, int n_in,
                              void* d_out, int out_size, void* d_ws, size_t ws_size,
                              hipStream_t stream) {
    (void)in_sizes; (void)n_in; (void)d_ws; (void)ws_size; (void)out_size;
    const float* batch = (const float*)d_in[0];
    const float* W1    = (const float*)d_in[1];
    const float* W2    = (const float*)d_in[2];
    float*       out   = (float*)d_out;

    dim3 grid(4, B_);      // 4 T-segments x 100 batch rows = 400 blocks
    dim3 block(256);       // 8 wave32 waves: 7 WMMA waves + staging helpers
    lif_fused_kernel<<<grid, block, 0, stream>>>(batch, W1, W2, out);
}
